// Het_Agg_49323404427454
// MI455X (gfx1250) — compile-verified, matrix operand order checked by reading, not run on hardware
//
#include <hip/hip_runtime.h>
#include <hip/hip_bf16.h>
#include <math.h>

typedef __attribute__((ext_vector_type(16))) _Float16 v16h;
typedef __attribute__((ext_vector_type(8)))  float    v8f;

#define D 256                 // embed dim
#define A_ROW_F 36            // A panel LDS row stride in floats (32 + 4 pad, bank-conflict free)
#define A_BUF_BYTES (128 * A_ROW_F * 4)   // one A double-buffer slab: 18432 B

__device__ __forceinline__ float warp_sum(float v) {
  for (int off = 16; off > 0; off >>= 1) v += __shfl_xor(v, off, 32);
  return v;
}

__device__ __forceinline__ void pack8(v16h& v, int base, float4 p, float4 q) {
  v[base + 0] = (_Float16)p.x; v[base + 1] = (_Float16)p.y;
  v[base + 2] = (_Float16)p.z; v[base + 3] = (_Float16)p.w;
  v[base + 4] = (_Float16)q.x; v[base + 5] = (_Float16)q.y;
  v[base + 6] = (_Float16)q.z; v[base + 7] = (_Float16)q.w;
}

// Issue async global->LDS copies for one 128-row x 32-float A panel k-step.
// 1024 16B chunks spread over 256 threads (4 async b128 per thread).
// Tracked by ASYNCcnt; caller sequences with s_wait_asynccnt + barrier.
__device__ __forceinline__ void stage_A_async(const float* __restrict__ X, int N, int K,
                                              int rbase, int s, char* Abuf, int tid) {
  #pragma unroll
  for (int j = 0; j < 4; ++j) {
    const int q    = tid + (j << 8);     // 0..1023
    const int row  = q >> 3;             // 0..127
    const int part = q & 7;              // 16B chunk within 32-float row window
    int gr = rbase + row;
    if (gr > N - 1) gr = N - 1;          // clamp (duplicate row; stores are guarded)
    const float* g = X + (size_t)gr * K + (s << 5) + (part << 2);
    unsigned int loff = (unsigned int)(size_t)(Abuf + (row * A_ROW_F + (part << 2)) * 4);
    asm volatile("global_load_async_to_lds_b128 %0, %1, off"
                 :: "v"(loff), "v"(g) : "memory");
  }
}

// Y[N x Dout] = relu(X[N x K] @ W^T + b), W row-major [Dout x K].
// Block: 256 threads (8 waves). Block tile: 128 rows x 64 cols.
// Wave w: rows [w*16, w*16+16), all 64 cols via 4 f32 accumulators sharing one A fragment.
// B slice (64 cols x K) staged once into LDS as pre-converted f16 WMMA fragments.
// A panel (128 x 32 f32) double-buffered in LDS via global_load_async_to_lds_b128,
// pipelined with s_wait_asynccnt + workgroup barrier.
__global__ void gemm_relu_wmma(const float* __restrict__ X,
                               const float* __restrict__ W,
                               const float* __restrict__ bias,
                               float* __restrict__ Y,
                               int N, int K, int Dout) {
  extern __shared__ __align__(64) char smem_raw[];
  _Float16* Bf   = (_Float16*)smem_raw;              // K*128 bytes of f16 fragments
  char*     Abase = smem_raw + (size_t)K * 128;      // 2 * A_BUF_BYTES

  const int tid  = threadIdx.x;
  const int wv   = tid >> 5;
  const int lane = tid & 31;
  const int r15  = lane & 15;
  const int half = lane >> 4;
  const int tq   = blockIdx.x;           // 64-col group
  const int rbase = blockIdx.y * 128;    // 128-row group

  // ---- Stage B fragments: (s, c, l) fragment = 16 f16 in WMMA B layout ----
  const int nSteps = K >> 5;
  const int F = nSteps << 7;             // nSteps * 4 coltiles * 32 lanes
  for (int f = tid; f < F; f += 256) {
    const int s   = f >> 7;
    const int rem = f & 127;
    const int c   = rem >> 5;
    const int l   = rem & 31;
    const int col = tq * 64 + c * 16 + (l & 15);
    // B[k][col] = W[col*K + k]; k runs: [s*32 + (l>>4)*8 .. +7] and [+16 .. +7]
    const float* wp = W + (size_t)col * K + (s << 5) + ((l >> 4) << 3);
    float4 g0 = *(const float4*)(wp);
    float4 g1 = *(const float4*)(wp + 4);
    float4 g2 = *(const float4*)(wp + 16);
    float4 g3 = *(const float4*)(wp + 20);
    v16h frag;
    pack8(frag, 0, g0, g1); pack8(frag, 8, g2, g3);
    *(v16h*)(Bf + (size_t)f * 16) = frag;
  }

  // ---- Async prologue: fill A buffer 0 ----
  stage_A_async(X, N, K, rbase, 0, Abase, tid);
  asm volatile("s_wait_asynccnt 0x0" ::: "memory");
  __syncthreads();

  v8f acc[4] = {{}, {}, {}, {}};
  for (int s = 0; s < nSteps; ++s) {
    char* cur = Abase + (s & 1) * A_BUF_BYTES;
    char* nxt = Abase + ((s & 1) ^ 1) * A_BUF_BYTES;
    if (s + 1 < nSteps) stage_A_async(X, N, K, rbase, s + 1, nxt, tid);

    // A fragment from LDS: row = wv*16 + r15, two 8-float runs per ISA layout
    const float* ap = (const float*)cur + (wv * 16 + r15) * A_ROW_F + half * 8;
    float4 a0 = *(const float4*)(ap);
    float4 a1 = *(const float4*)(ap + 4);
    float4 a2 = *(const float4*)(ap + 16);
    float4 a3 = *(const float4*)(ap + 20);
    v16h av;
    pack8(av, 0, a0, a1); pack8(av, 8, a2, a3);

    const _Float16* bs = Bf + (((size_t)s << 7) + lane) * 16;
    v16h b0 = *(const v16h*)(bs);
    v16h b1 = *(const v16h*)(bs + 32 * 16);
    v16h b2 = *(const v16h*)(bs + 64 * 16);
    v16h b3 = *(const v16h*)(bs + 96 * 16);

    acc[0] = __builtin_amdgcn_wmma_f32_16x16x32_f16(false, av, false, b0, (short)0, acc[0], false, false);
    acc[1] = __builtin_amdgcn_wmma_f32_16x16x32_f16(false, av, false, b1, (short)0, acc[1], false, false);
    acc[2] = __builtin_amdgcn_wmma_f32_16x16x32_f16(false, av, false, b2, (short)0, acc[2], false, false);
    acc[3] = __builtin_amdgcn_wmma_f32_16x16x32_f16(false, av, false, b3, (short)0, acc[3], false, false);

    // next A buffer landed + all waves done reading cur before it is overwritten
    asm volatile("s_wait_asynccnt 0x0" ::: "memory");
    __syncthreads();
  }

  // ---- Epilogue: bias + relu + guarded store (D layout: acc[r] -> M=r+8*half, N=r15) ----
  const int mbase = rbase + wv * 16;
  #pragma unroll
  for (int c = 0; c < 4; ++c) {
    const int n  = tq * 64 + c * 16 + r15;
    const float bn = bias[n];
    #pragma unroll
    for (int r = 0; r < 8; ++r) {
      const int m = mbase + r + half * 8;
      if (m < N) {
        float v = acc[c][r] + bn;
        Y[(size_t)m * Dout + n] = v > 0.f ? v : 0.f;
      }
    }
  }
}

// One wave per edge: Aggr[src] += Xt[tgt]; Deg[src] += 1  (HW f32 atomics)
__global__ void scatter_kernel(const float* __restrict__ Xt,
                               const long long* __restrict__ edges,
                               long long E,
                               float* __restrict__ Ag,
                               float* __restrict__ Dg) {
  long long e = (long long)blockIdx.x * (blockDim.x >> 5) + (threadIdx.x >> 5);
  if (e >= E) return;
  const int lane = threadIdx.x & 31;
  const int src = (int)edges[e];
  const int tgt = (int)edges[E + e];
  const float4* sp = (const float4*)(Xt + (size_t)tgt * D) + lane * 2;
  float* dp = Ag + (size_t)src * D + lane * 8;
  __builtin_prefetch(dp, 1, 1);
  float4 v0 = sp[0], v1 = sp[1];
  __hip_atomic_fetch_add(dp + 0, v0.x, __ATOMIC_RELAXED, __HIP_MEMORY_SCOPE_AGENT);
  __hip_atomic_fetch_add(dp + 1, v0.y, __ATOMIC_RELAXED, __HIP_MEMORY_SCOPE_AGENT);
  __hip_atomic_fetch_add(dp + 2, v0.z, __ATOMIC_RELAXED, __HIP_MEMORY_SCOPE_AGENT);
  __hip_atomic_fetch_add(dp + 3, v0.w, __ATOMIC_RELAXED, __HIP_MEMORY_SCOPE_AGENT);
  __hip_atomic_fetch_add(dp + 4, v1.x, __ATOMIC_RELAXED, __HIP_MEMORY_SCOPE_AGENT);
  __hip_atomic_fetch_add(dp + 5, v1.y, __ATOMIC_RELAXED, __HIP_MEMORY_SCOPE_AGENT);
  __hip_atomic_fetch_add(dp + 6, v1.z, __ATOMIC_RELAXED, __HIP_MEMORY_SCOPE_AGENT);
  __hip_atomic_fetch_add(dp + 7, v1.w, __ATOMIC_RELAXED, __HIP_MEMORY_SCOPE_AGENT);
  if (lane == 0)
    __hip_atomic_fetch_add(Dg + src, 1.0f, __ATOMIC_RELAXED, __HIP_MEMORY_SCOPE_AGENT);
}

// One wave per node: degree-normalize, attention softmax-of-2, build [x_node, combined]
__global__ void combine_kernel(const float* __restrict__ Ua, const float* __restrict__ Ma,
                               const float* __restrict__ DgU, const float* __restrict__ DgM,
                               const float* __restrict__ Xn, const float* __restrict__ attn,
                               float* __restrict__ C2, int N) {
  int n = blockIdx.x * (blockDim.x >> 5) + (threadIdx.x >> 5);
  if (n >= N) return;
  const int lane = threadIdx.x & 31;
  const float iu = 1.f / fmaxf(DgU[n], 1.f);
  const float im = 1.f / fmaxf(DgM[n], 1.f);
  const float* uap = Ua + (size_t)n * D + lane * 8;
  const float* map = Ma + (size_t)n * D + lane * 8;
  const float* xp  = Xn + (size_t)n * D + lane * 8;
  const float* a0  = attn + lane * 8;
  const float* a1  = attn + D + lane * 8;
  float ua[8], ma[8], xv[8];
  float pu = 0.f, pm = 0.f, px = 0.f;
  #pragma unroll
  for (int j = 0; j < 8; ++j) {
    ua[j] = uap[j] * iu;
    ma[j] = map[j] * im;
    xv[j] = xp[j];
    pu += ua[j] * a0[j];
    pm += ma[j] * a0[j];
    px += xv[j] * a1[j];
  }
  pu = warp_sum(pu); pm = warp_sum(pm); px = warp_sum(px);
  float su = pu + px, sm = pm + px;
  su = su > 0.f ? su : 0.01f * su;
  sm = sm > 0.f ? sm : 0.01f * sm;
  su = __expf(su); sm = __expf(sm);
  const float w = su / (su + sm);
  float* c0 = C2 + (size_t)n * (2 * D) + lane * 8;
  float* c1 = c0 + D;
  #pragma unroll
  for (int j = 0; j < 8; ++j) {
    c0[j] = xv[j];
    c1[j] = w * ua[j] + (1.f - w) * ma[j];
  }
}

// One wave per node: row L2 normalize
__global__ void normalize_kernel(const float* __restrict__ H, float* __restrict__ O, int N) {
  int n = blockIdx.x * (blockDim.x >> 5) + (threadIdx.x >> 5);
  if (n >= N) return;
  const int lane = threadIdx.x & 31;
  const float* hp = H + (size_t)n * D + lane * 8;
  float h[8]; float ss = 0.f;
  #pragma unroll
  for (int j = 0; j < 8; ++j) { h[j] = hp[j]; ss += h[j] * h[j]; }
  ss = warp_sum(ss);
  const float scale = 1.f / fmaxf(sqrtf(ss), 1e-12f);
  float* op = O + (size_t)n * D + lane * 8;
  #pragma unroll
  for (int j = 0; j < 8; ++j) op[j] = h[j] * scale;
}

extern "C" void kernel_launch(void* const* d_in, const int* in_sizes, int n_in,
                              void* d_out, int out_size, void* d_ws, size_t ws_size,
                              hipStream_t stream) {
  const float*     x_u     = (const float*)d_in[0];
  const float*     x_m     = (const float*)d_in[1];
  const float*     x_node  = (const float*)d_in[2];
  const long long* edges_u = (const long long*)d_in[3];
  const long long* edges_m = (const long long*)d_in[4];
  // d_in[5] = num_node scalar (unused; derived from x_node size)
  const float*     Wu      = (const float*)d_in[6];
  const float*     bu      = (const float*)d_in[7];
  const float*     Wm      = (const float*)d_in[8];
  const float*     bm      = (const float*)d_in[9];
  const float*     attn    = (const float*)d_in[10];
  const float*     Wl      = (const float*)d_in[11];
  const float*     bl      = (const float*)d_in[12];
  float*           out     = (float*)d_out;

  const int Nu = in_sizes[0] / D;
  const int Nm = in_sizes[1] / D;
  const int Nn = in_sizes[2] / D;
  const long long Eu = in_sizes[3] / 2;
  const long long Em = in_sizes[4] / 2;

  auto alignup = [](size_t x) { return (x + 255) & ~(size_t)255; };
  char* ws = (char*)d_ws;
  size_t off = 0;
  float* Xt_u = (float*)(ws + off); off += alignup((size_t)Nu * D * sizeof(float));
  float* Xt_m = (float*)(ws + off); off += alignup((size_t)Nm * D * sizeof(float));
  float* Ag_u = (float*)(ws + off); off += alignup((size_t)Nn * D * sizeof(float));
  float* Ag_m = (float*)(ws + off); off += alignup((size_t)Nn * D * sizeof(float));
  float* Dg_u = (float*)(ws + off); off += alignup((size_t)Nn * sizeof(float));
  float* Dg_m = (float*)(ws + off); off += alignup((size_t)Nn * sizeof(float));
  float* C2   = (float*)(ws + off); off += alignup((size_t)Nn * 2 * D * sizeof(float));
  float* H    = Xt_u;   // Xt_u dead after scatter phase; reuse for final GEMM output

  hipMemsetAsync(Ag_u, 0, (size_t)Nn * D * sizeof(float), stream);
  hipMemsetAsync(Ag_m, 0, (size_t)Nn * D * sizeof(float), stream);
  hipMemsetAsync(Dg_u, 0, (size_t)Nn * sizeof(float), stream);
  hipMemsetAsync(Dg_m, 0, (size_t)Nn * sizeof(float), stream);

  const dim3 blk(256);  // 8 waves

  // Phase 1: x_t = relu(x @ W^T + b)  (K=256, Dout=256)
  {
    const size_t shmem = (size_t)D * 128 + 2 * A_BUF_BYTES;  // 69632 B
    dim3 grd(D / 64, (unsigned)((Nu + 127) / 128));
    gemm_relu_wmma<<<grd, blk, shmem, stream>>>(x_u, Wu, bu, Xt_u, Nu, D, D);
    dim3 grd2(D / 64, (unsigned)((Nm + 127) / 128));
    gemm_relu_wmma<<<grd2, blk, shmem, stream>>>(x_m, Wm, bm, Xt_m, Nm, D, D);
  }

  // Phase 2: segment-sum + degree via HW f32 atomics
  scatter_kernel<<<dim3((unsigned)((Eu + 7) / 8)), blk, 0, stream>>>(Xt_u, edges_u, Eu, Ag_u, Dg_u);
  scatter_kernel<<<dim3((unsigned)((Em + 7) / 8)), blk, 0, stream>>>(Xt_m, edges_m, Em, Ag_m, Dg_m);

  // Phase 3: attention mix -> [x_node, combined]  (N x 512)
  combine_kernel<<<dim3((Nn + 7) / 8), blk, 0, stream>>>(Ag_u, Ag_m, Dg_u, Dg_m,
                                                         x_node, attn, C2, Nn);

  // Phase 4: h = relu(C2 @ Wl^T + bl)  (K=512, Dout=256)
  {
    const size_t shmem = (size_t)(2 * D) * 128 + 2 * A_BUF_BYTES;  // 102400 B
    dim3 grd(D / 64, (unsigned)((Nn + 127) / 128));
    gemm_relu_wmma<<<grd, blk, shmem, stream>>>(C2, Wl, bl, H, Nn, 2 * D, D);
  }

  // Phase 5: row L2 normalize -> output
  normalize_kernel<<<dim3((Nn + 7) / 8), blk, 0, stream>>>(H, out, Nn);
}
